// DynamicSnakeConv2D_21663815041546
// MI455X (gfx1250) — matrix operand order repaired and checked by reading, not compiled, not run on hardware
//
#include <hip/hip_runtime.h>
#include <hip/hip_bf16.h>
#include <math.h>

typedef __attribute__((ext_vector_type(16))) _Float16 v16h;
typedef __attribute__((ext_vector_type(8)))  _Float16 v8h;
typedef __attribute__((ext_vector_type(8)))  float    v8f;

#define HH 128
#define WWD 128
#define CC 128
#define FF 256
#define KC 1152      // 9*128 reduction length
#define KSTEPS 36    // 1152 / 32
#define NTILES 16    // 256 / 16
#define MT 32        // M tile (pixels per block)
#define LDA 1160     // padded A row length in f16 elements (2320B, 16B aligned)

// ---------------------------------------------------------------------------
// Kernel 0: repack GEMM weights f32 [1152,256] -> f16, pre-swizzled into the
// per-lane B-fragment layout for v_wmma_f32_16x16x32_f16.
// B fragment (32x16): column n = lane&15, k = kstep*32 + (lane>>4)*8 + (half&7) + (half>>3)*16.
// Stored as Bw[kstep][ntile][lane][16 halves] -> 32 contiguous bytes per lane.
// ---------------------------------------------------------------------------
__global__ void repack_weights(const float* __restrict__ kw,
                               _Float16* __restrict__ Bw) {
    int gid   = blockIdx.x * blockDim.x + threadIdx.x;  // 0 .. 294911
    int h     = gid & 15;
    int lane  = (gid >> 4) & 31;
    int ntile = (gid >> 9) & 15;
    int kstep = gid >> 13;
    int k = kstep * 32 + (lane >> 4) * 8 + (h & 7) + (h >> 3) * 16;
    int n = ntile * 16 + (lane & 15);
    Bw[gid] = (_Float16)kw[(size_t)k * FF + n];
}

// ---------------------------------------------------------------------------
// Kernel 1: per-pixel offset (18ch) + modulation (9ch) 3x3 SAME conv in fp32,
// then clip/sigmoid prep -> params[pixel][9] = {gy, gx, sigmoid(mod)}.
// Kept fp32 because these values feed floor() / bilinear weights.
// ---------------------------------------------------------------------------
__global__ void offmod(const float* __restrict__ x,
                       const float* __restrict__ ow, const float* __restrict__ ob,
                       const float* __restrict__ mw, const float* __restrict__ mb,
                       float* __restrict__ params) {
    int p = blockIdx.x * blockDim.x + threadIdx.x;  // 0..65535 pixel id
    int b = p >> 14;
    int h = (p >> 7) & 127;
    int w = p & 127;

    float acc[27];
#pragma unroll
    for (int o = 0; o < 18; ++o) acc[o] = ob[o];
#pragma unroll
    for (int o = 0; o < 9; ++o) acc[18 + o] = mb[o];

    for (int kh = 0; kh < 3; ++kh) {
        int hh = h + kh - 1;
        if (hh < 0 || hh >= HH) continue;
        for (int kw2 = 0; kw2 < 3; ++kw2) {
            int ww = w + kw2 - 1;
            if (ww < 0 || ww >= WWD) continue;
            const float* xp = x + (((size_t)(b * HH + hh) * WWD + ww) * CC);
            int wbase = (kh * 3 + kw2) * CC;
            for (int c = 0; c < CC; ++c) {
                float xv = xp[c];
                const float* wo = ow + (size_t)(wbase + c) * 18;
                const float* wm = mw + (size_t)(wbase + c) * 9;
#pragma unroll
                for (int o = 0; o < 18; ++o) acc[o] = fmaf(xv, wo[o], acc[o]);
#pragma unroll
                for (int o = 0; o < 9; ++o) acc[18 + o] = fmaf(xv, wm[o], acc[18 + o]);
            }
        }
    }

#pragma unroll
    for (int kk = 0; kk < 9; ++kk) {
        float ky = (float)(kk / 3 - 1);
        float kx = (float)(kk % 3 - 1);
        float gy = ky + acc[2 * kk];
        float gx = kx + acc[2 * kk + 1];
        gy = fminf(fmaxf(gy, 0.0f), (float)(HH - 1));
        gx = fminf(fmaxf(gx, 0.0f), (float)(WWD - 1));
        float mv = 1.0f / (1.0f + expf(-acc[18 + kk]));
        float* pp = params + ((size_t)p * 9 + kk) * 3;
        pp[0] = gy; pp[1] = gx; pp[2] = mv;
    }
}

// ---------------------------------------------------------------------------
// Kernel 2: main fused kernel. Block = 512 threads = 16 waves.
// Handles 32 pixels (M tile) x 256 outputs (16 N tiles, one per wave).
//   Phase 1: 288 threads compute bilinear coords/weights -> LDS.
//   Phase 2: all threads build 32x1152 f16 A tile (modulated samples) in LDS.
//   Phase 3: each wave runs 36 K-steps; each B fragment is loaded once and
//            reused by TWO independent wmma accumulator chains (rows 0-15 and
//            16-31) -> halved B traffic from L2 + 2x WMMA ILP.
// ---------------------------------------------------------------------------
__global__ void __launch_bounds__(512, 1) deform_gemm(
        const float* __restrict__ x,
        const float* __restrict__ params,
        const _Float16* __restrict__ Bw,
        const float* __restrict__ bias,
        float* __restrict__ out) {
    __shared__ __align__(16) _Float16 As[MT * LDA];          // 74240 B
    __shared__ int   sy0[MT * 9], sx0[MT * 9], sy1[MT * 9], sx1[MT * 9];
    __shared__ float sw00[MT * 9], sw01[MT * 9], sw10[MT * 9], sw11[MT * 9],
                     smod[MT * 9];

    const int tid = threadIdx.x;
    const int p0  = blockIdx.x * MT;

    // Phase 1: sampling parameters for 32 pixels x 9 taps
    if (tid < MT * 9) {
        int m  = tid / 9;
        int kk = tid - m * 9;
        const float* pp = params + ((size_t)(p0 + m) * 9 + kk) * 3;
        float gy = pp[0], gx = pp[1], mv = pp[2];
        int y0 = (int)floorf(gy);
        int x0 = (int)floorf(gx);
        int y1 = min(y0 + 1, HH - 1);
        int x1 = min(x0 + 1, WWD - 1);
        float wy1 = gy - (float)y0, wy0 = 1.0f - wy1;
        float wx1 = gx - (float)x0, wx0 = 1.0f - wx1;
        sy0[tid] = y0; sx0[tid] = x0; sy1[tid] = y1; sx1[tid] = x1;
        sw00[tid] = wy0 * wx0; sw01[tid] = wy0 * wx1;
        sw10[tid] = wy1 * wx0; sw11[tid] = wy1 * wx1;
        smod[tid] = mv;
    }
    __syncthreads();

    // Phase 2: build A tile (32 pixels x 1152 K), K order = tap*128 + channel
#pragma unroll 4
    for (int t = 0; t < (MT * KC) / 512; ++t) {   // 72 iterations
        int e  = t * 512 + tid;                    // 0..36863
        int m  = e / KC;
        int r  = e - m * KC;
        int kk = r >> 7;
        int c  = r & 127;
        int pi = p0 + m;
        int b  = pi >> 14;
        int s  = m * 9 + kk;
        const float* xb = x + ((size_t)b * HH * WWD * CC);
        float v00 = xb[((size_t)(sy0[s] * WWD + sx0[s]) * CC) + c];
        float v01 = xb[((size_t)(sy0[s] * WWD + sx1[s]) * CC) + c];
        float v10 = xb[((size_t)(sy1[s] * WWD + sx0[s]) * CC) + c];
        float v11 = xb[((size_t)(sy1[s] * WWD + sx1[s]) * CC) + c];
        float v = sw00[s] * v00 + sw01[s] * v01 + sw10[s] * v10 + sw11[s] * v11;
        v *= smod[s];
        As[m * LDA + r] = (_Float16)v;
    }
    __syncthreads();

    // Phase 3: WMMA. wave wv owns output columns [wv*16, wv*16+16),
    // two M sub-tiles (rows 0-15 / 16-31) sharing each B fragment.
    const int wv   = tid >> 5;
    const int lane = tid & 31;
    const int mrow = lane & 15;
    const int khi  = lane >> 4;

    v8f acc0 = {};
    v8f acc1 = {};
    for (int ks = 0; ks < KSTEPS; ++ks) {
        // A fragment per ISA 16-bit A layout:
        // lanes 0-15: halves 0-7 = K+0..7,  halves 8-15 = K+16..23
        // lanes 16-31: halves 0-7 = K+8..15, halves 8-15 = K+24..31
        union { v16h v; v8h h2[2]; } a0, a1;
        const _Float16* ap0 = &As[mrow * LDA + ks * 32 + khi * 8];
        const _Float16* ap1 = ap0 + 16 * LDA;
        a0.h2[0] = *(const v8h*)(ap0);        // ds_load_b128
        a0.h2[1] = *(const v8h*)(ap0 + 16);
        a1.h2[0] = *(const v8h*)(ap1);
        a1.h2[1] = *(const v8h*)(ap1 + 16);
        // B fragment: 32 contiguous bytes per lane from pre-swizzled Bw
        v16h bfrag = *(const v16h*)(Bw + (((size_t)ks * NTILES + wv) * 32 + lane) * 16);
        acc0 = __builtin_amdgcn_wmma_f32_16x16x32_f16(
            false, a0.v, false, bfrag, (short)0, acc0, false, false);
        acc1 = __builtin_amdgcn_wmma_f32_16x16x32_f16(
            false, a1.v, false, bfrag, (short)0, acc1, false, false);
    }

    // C/D layout: VGPR j -> M = j + 8*(lane>>4), N = lane&15
    int n = wv * 16 + mrow;
    float bv = bias[n];
#pragma unroll
    for (int j = 0; j < 8; ++j) {
        int m = j + khi * 8;
        out[(size_t)(p0 + m) * FF + n]      = acc0[j] + bv;
        out[(size_t)(p0 + 16 + m) * FF + n] = acc1[j] + bv;
    }
}

// ---------------------------------------------------------------------------
extern "C" void kernel_launch(void* const* d_in, const int* in_sizes, int n_in,
                              void* d_out, int out_size, void* d_ws, size_t ws_size,
                              hipStream_t stream) {
    const float* x    = (const float*)d_in[0];
    const float* ow   = (const float*)d_in[1];
    const float* ob   = (const float*)d_in[2];
    const float* mw   = (const float*)d_in[3];
    const float* mb   = (const float*)d_in[4];
    const float* kw   = (const float*)d_in[5];
    const float* bias = (const float*)d_in[6];
    float* out = (float*)d_out;

    char* ws = (char*)d_ws;
    _Float16* Bw  = (_Float16*)ws;              // 36*16*32*16 f16 = 589824 B
    float*    prm = (float*)(ws + 589824);      // 65536*9*3 f32   = 7077888 B

    repack_weights<<<1152, 256, 0, stream>>>(kw, Bw);          // 294912 elems
    offmod<<<256, 256, 0, stream>>>(x, ow, ob, mw, mb, prm);   // 65536 pixels
    deform_gemm<<<65536 / MT, 512, 0, stream>>>(x, prm, Bw, bias, out);
}
